// DebuggableInvariantPointAttention_22634477650374
// MI455X (gfx1250) — compile-verified, matrix operand order checked by reading, not run on hardware
//
#include <hip/hip_runtime.h>
#include <hip/hip_bf16.h>

// ---------------------------------------------------------------------------
// DebuggableInvariantPointAttention for MI455X (gfx1250, wave32, WMMA + TDM).
// All GEMMs run on v_wmma_f32_16x16x32_bf16 (bf16 in, f32 accumulate).
// The op is HBM-bound on z (268 MB read twice at 23.3 TB/s); the two kernels
// that stream z stage tiles into LDS with the Tensor Data Mover
// (tensor_load_to_lds + s_wait_tensorcnt) and share them across 8 waves.
// ---------------------------------------------------------------------------

typedef __attribute__((ext_vector_type(16))) __bf16 v16bf;
typedef __attribute__((ext_vector_type(8)))  float  v8f;
typedef __attribute__((ext_vector_type(4)))  unsigned int u32x4;
typedef __attribute__((ext_vector_type(8)))  int    i32x8;
typedef __attribute__((ext_vector_type(4)))  int    i32x4;

// Problem constants (match reference)
constexpr int   BB  = 2;
constexpr int   NN  = 512;
constexpr int   CS  = 384;
constexpr int   CZ  = 128;
constexpr int   CH  = 16;
constexpr int   HH  = 12;
constexpr float INF = 100000.0f;

// ---------------------------------------------------------------------------
// WMMA fragment helpers (ISA cdna5 CDNA5 ISA §7.12.2, wave32 layouts)
// A 16x32 bf16: lane L<16 holds M=L, K={0..7,16..23}; L>=16: K={8..15,24..31}.
// B 32x16 bf16: column-major mirror (lane&15 = N, same K striping).
// C/D 16x16 f32: VGPR r -> M = r + 8*(lane>=16), N = lane&15.
// ---------------------------------------------------------------------------
template <typename F>
__device__ __forceinline__ v16bf make_frag(F f) {
  const int lane = threadIdx.x & 31;
  const int mn   = lane & 15;
  const int half = lane >> 4;
  v16bf r;
#pragma unroll
  for (int e = 0; e < 16; ++e) {
    const int k = half * 8 + ((e < 8) ? e : (e + 8));
    r[e] = (__bf16)f(mn, k);
  }
  return r;
}

template <typename F>
__device__ __forceinline__ void store_acc(const v8f& c, F f) {
  const int lane = threadIdx.x & 31;
  const int n    = lane & 15;
  const int half = lane >> 4;
#pragma unroll
  for (int r = 0; r < 8; ++r) f(r + half * 8, n, c[r]);
}

__device__ __forceinline__ v8f wmma_bf16(v16bf a, v16bf b, v8f c) {
  // 8 args: (neg_a, A, neg_b, B, c_mod, C, reuse_a, reuse_b)
  return __builtin_amdgcn_wmma_f32_16x16x32_bf16(false, a, false, b,
                                                 (short)0, c, false, false);
}

// ---------------------------------------------------------------------------
// TDM: DMA a 2D f32 tile (tile_h rows x tile_w elems, row stride in elems)
// from global memory into LDS at byte offset lds_off.  Descriptor layout per
// CDNA5 ISA: group0 = {count, lds_addr, global_addr,
// type=2}; group1 = {data_size=4B, tensor dims, tile dims, dim0 stride}.
// Tracked with TENSORcnt.
// ---------------------------------------------------------------------------
__device__ __forceinline__ void tdm_load_2d(unsigned lds_off,
                                            const void* gptr,
                                            unsigned tile_w_elems,
                                            unsigned tile_h_rows,
                                            unsigned row_stride_elems) {
  const unsigned long long ga = (unsigned long long)gptr;
  u32x4 g0;
  g0[0] = 1u;                                            // count=1, user mode
  g0[1] = lds_off;                                       // lds_addr (bytes)
  g0[2] = (unsigned)(ga & 0xffffffffu);                  // global_addr[31:0]
  g0[3] = (unsigned)((ga >> 32) & 0x1ffffffu) | (2u << 30); // addr[56:32]|type=2
  i32x8 g1;
  g1[0] = (int)(2u << 16);                               // data_size=4B, no mcast/pad
  g1[1] = (int)(tile_w_elems << 16);                     // tensor_dim0[15:0]
  g1[2] = (int)(tile_h_rows << 16);                      // dim0 hi | tensor_dim1[15:0]
  g1[3] = (int)(tile_w_elems << 16);                     // dim1 hi | tile_dim0
  g1[4] = (int)tile_h_rows;                              // tile_dim1 | tile_dim2=0
  g1[5] = (int)row_stride_elems;                         // tensor_dim0_stride[31:0]
  g1[6] = 0;
  g1[7] = 0;
  i32x4 gz = {0, 0, 0, 0};                               // 2D: groups 2/3 unused
#if __clang_major__ >= 23
  i32x8 gz8 = {0, 0, 0, 0, 0, 0, 0, 0};
  __builtin_amdgcn_tensor_load_to_lds(g0, g1, gz, gz, gz8, 0);
#else
  __builtin_amdgcn_tensor_load_to_lds(g0, g1, gz, gz, 0);
#endif
}

// ---------------------------------------------------------------------------
// K1: generic bf16 GEMM  C[M,N] = A[M,K] @ W[K,N]   (dims % 16 == 0, K % 32 == 0)
// one wave (block of 32) per 16x16 output tile
// ---------------------------------------------------------------------------
__global__ void gemm_proj_kernel(const float* __restrict__ A,
                                 const float* __restrict__ W,
                                 float* __restrict__ C,
                                 int Ncols, int K, int ldA, int ldW, int ldC) {
  const int tiles_n = Ncols / 16;
  const int tn = blockIdx.x % tiles_n;
  const int tm = blockIdx.x / tiles_n;
  const float* Ab = A + (long)tm * 16 * ldA;
  const float* Wb = W + tn * 16;
  v8f acc = {};
  for (int k0 = 0; k0 < K; k0 += 32) {
    v16bf af = make_frag([&](int m, int k) { return Ab[(long)m * ldA + k0 + k]; });
    v16bf bf = make_frag([&](int n, int k) { return Wb[(long)(k0 + k) * ldW + n]; });
    acc = wmma_bf16(af, bf, acc);
  }
  float* Cb = C + (long)tm * 16 * ldC + tn * 16;
  store_acc(acc, [&](int m, int n, float v) { Cb[(long)m * ldC + n] = v; });
}

// ---------------------------------------------------------------------------
// K2: b = z @ Wb, stored transposed+scaled as bt[b,h,i,j] = b*sqrt(0.5);
// also emits a_b = bt * (mask_i*mask_j).
// 256 threads = 8 waves; each block owns 128 consecutive flat z rows.
// Wave 0 TDM-stages a 128x32 strided f32 tile of z into LDS per K step.
// ---------------------------------------------------------------------------
__global__ void bias_kernel(const float* __restrict__ z,
                            const float* __restrict__ Wb,
                            const float* __restrict__ mask,
                            float* __restrict__ bt_ws,
                            float* __restrict__ a_b_out) {
  __shared__ float zb[128 * 32];                         // 16 KB staged K-tile
  const int  wave = threadIdx.x >> 5;
  const long row0 = (long)blockIdx.x * 128;
  v8f acc = {};
  for (int k0 = 0; k0 < CZ; k0 += 32) {
    __syncthreads();                                     // previous tile consumed
    if (wave == 0) {
      tdm_load_2d(0, z + row0 * CZ + k0, 32, 128, CZ);
      __builtin_amdgcn_s_wait_tensorcnt(0);
    }
    __syncthreads();                                     // tile visible to all waves
    v16bf af = make_frag([&](int m, int k) { return zb[(wave * 16 + m) * 32 + k]; });
    v16bf bw = make_frag([&](int n, int k) { return (n < HH) ? Wb[(long)(k0 + k) * HH + n] : 0.0f; });
    acc = wmma_bf16(af, bw, acc);
  }
  const long wrow0 = row0 + wave * 16;
  store_acc(acc, [&](int m, int h, float v) {
    if (h < HH) {
      const long row = wrow0 + m;
      const int  j   = (int)(row % NN);
      const long t   = row / NN;
      const int  i   = (int)(t % NN);
      const int  bb  = (int)(t / NN);
      const float btv = v * 0.70710678118654752f;        // * sqrt(0.5)
      const long idx  = (((long)bb * HH + h) * NN + i) * NN + j;
      bt_ws[idx]   = btv;
      a_b_out[idx] = btv * mask[bb * NN + i] * mask[bb * NN + j];
    }
  });
}

// ---------------------------------------------------------------------------
// K3: scores = q.k^T * sqrt(1/32); a_sd = scores*sq; logits (in-place over bt)
//     = scores + bt + INF*(sq-1).   K = C_H = 16, zero-padded to 32.
// ---------------------------------------------------------------------------
__global__ void logits_kernel(const float* __restrict__ qws,
                              const float* __restrict__ kvws,
                              const float* __restrict__ mask,
                              float* __restrict__ bt_ws,
                              float* __restrict__ a_sd_out) {
  int id = blockIdx.x;
  const int jt = id & 31; id >>= 5;
  const int it = id & 31; id >>= 5;
  const int h  = id % HH;
  const int bb = id / HH;
  const float* Qb = qws  + ((long)(bb * NN + it * 16)) * (HH * CH) + h * CH;
  const float* Kb = kvws + ((long)(bb * NN + jt * 16)) * (2 * HH * CH) + h * 2 * CH;
  v8f acc = {};
  v16bf af = make_frag([&](int m, int k) { return (k < CH) ? Qb[(long)m * (HH * CH) + k] : 0.0f; });
  v16bf bf = make_frag([&](int n, int k) { return (k < CH) ? Kb[(long)n * (2 * HH * CH) + k] : 0.0f; });
  acc = wmma_bf16(af, bf, acc);
  const float scale = 0.17677669529663687f;              // sqrt(1/(2*C_H))
  const long base = (((long)bb * HH + h) * NN + it * 16) * NN + jt * 16;
  store_acc(acc, [&](int m, int n, float v) {
    const int i = it * 16 + m, j = jt * 16 + n;
    const float sq = mask[bb * NN + i] * mask[bb * NN + j];
    const float sc = v * scale;
    const long idx = base + (long)m * NN + n;
    a_sd_out[idx] = sc * sq;
    bt_ws[idx]    = sc + bt_ws[idx] + INF * (sq - 1.0f);
  });
}

// ---------------------------------------------------------------------------
// K4: row softmax over j (512). One wave per row, 16 elems/lane.
// ---------------------------------------------------------------------------
__global__ void softmax_kernel(const float* __restrict__ logits,
                               float* __restrict__ a_out) {
  const int wave = blockIdx.x * (blockDim.x >> 5) + (threadIdx.x >> 5);
  const int lane = threadIdx.x & 31;
  const float* row = logits + (long)wave * NN;
  float vals[16];
  float mx = -3.4e38f;
#pragma unroll
  for (int t = 0; t < 16; ++t) { vals[t] = row[lane + 32 * t]; mx = fmaxf(mx, vals[t]); }
#pragma unroll
  for (int s = 16; s >= 1; s >>= 1) mx = fmaxf(mx, __shfl_xor(mx, s, 32));
  float sum = 0.0f;
#pragma unroll
  for (int t = 0; t < 16; ++t) { vals[t] = __expf(vals[t] - mx); sum += vals[t]; }
#pragma unroll
  for (int s = 16; s >= 1; s >>= 1) sum += __shfl_xor(sum, s, 32);
  const float inv = 1.0f / sum;
  float* orow = a_out + (long)wave * NN;
#pragma unroll
  for (int t = 0; t < 16; ++t) orow[lane + 32 * t] = vals[t] * inv;
}

// ---------------------------------------------------------------------------
// K5: o[b,i,h,:] = sum_j a[b,h,i,j] * v[b,j,h,:]   (per (b,h): 512x512 @ 512x16)
// ---------------------------------------------------------------------------
__global__ void av_kernel(const float* __restrict__ a,
                          const float* __restrict__ kvws,
                          float* __restrict__ ows) {
  int id = blockIdx.x;
  const int it = id & 31; id >>= 5;
  const int h  = id % HH;
  const int bb = id / HH;
  const float* Arow = a + (((long)bb * HH + h) * NN + it * 16) * NN;
  const float* Vb   = kvws + ((long)bb * NN) * (2 * HH * CH) + h * 2 * CH + CH;
  v8f acc = {};
  for (int k0 = 0; k0 < NN; k0 += 32) {
    __builtin_prefetch(Arow + k0 + 128, 0, 1);           // global_prefetch_b8
    v16bf af = make_frag([&](int m, int k) { return Arow[(long)m * NN + k0 + k]; });
    v16bf bf = make_frag([&](int n, int k) { return Vb[(long)(k0 + k) * (2 * HH * CH) + n]; });
    acc = wmma_bf16(af, bf, acc);
  }
  float* Cb = ows + ((long)(bb * NN + it * 16)) * (HH * CH) + h * CH;
  store_acc(acc, [&](int m, int n, float v) { Cb[(long)m * (HH * CH) + n] = v; });
}

// ---------------------------------------------------------------------------
// K6: o_pair[b,i,h,:] = sum_j a[b,h,i,j] * z[b,i,j,:]
//     (per (b,i): [12x512] @ [512x128], M padded to 16)
// One workgroup (8 waves) per (b,i); wave w owns column tile w of C_Z.
// Wave 0 TDM-stages the contiguous 32x128 K-slab of z[b,i] into LDS.
// ---------------------------------------------------------------------------
__global__ void az_kernel(const float* __restrict__ a,
                          const float* __restrict__ z,
                          float* __restrict__ opws) {
  __shared__ float zbuf[32 * CZ];                        // 16 KB staged K-tile
  const int wave = threadIdx.x >> 5;
  int id = blockIdx.x;
  const int i  = id % NN;
  const int bb = id / NN;
  const float* Ab = a + (((long)bb * HH) * NN + i) * NN; // A[h][j], row stride NN*NN
  const float* Zb = z + (((long)bb * NN + i) * NN) * CZ; // z[b,i] slab (512x128)
  v8f acc = {};
  for (int k0 = 0; k0 < NN; k0 += 32) {
    __syncthreads();
    if (wave == 0) {
      tdm_load_2d(0, Zb + (long)k0 * CZ, CZ, 32, CZ);    // contiguous 16 KB slab
      __builtin_amdgcn_s_wait_tensorcnt(0);
    }
    __syncthreads();
    v16bf af = make_frag([&](int m, int k) {
      return (m < HH) ? Ab[(long)m * NN * NN + k0 + k] : 0.0f;
    });
    v16bf bf = make_frag([&](int n, int k) { return zbuf[k * CZ + wave * 16 + n]; });
    acc = wmma_bf16(af, bf, acc);
  }
  float* Cb = opws + ((long)(bb * NN + i)) * (HH * CZ) + wave * 16;
  store_acc(acc, [&](int m, int n, float v) {
    if (m < HH) Cb[(long)m * CZ + n] = v;
  });
}

// ---------------------------------------------------------------------------
// K7: s_out = concat(o, o_pair) @ Wout + bout   (1024 x 1728 @ 1728 x 384)
// ---------------------------------------------------------------------------
__global__ void out_kernel(const float* __restrict__ ows,
                           const float* __restrict__ opws,
                           const float* __restrict__ Wout,
                           const float* __restrict__ bout,
                           float* __restrict__ s_out) {
  const int tn = blockIdx.x % (CS / 16);
  const int tm = blockIdx.x / (CS / 16);
  const float* Ob = ows  + (long)tm * 16 * (HH * CH);
  const float* Pb = opws + (long)tm * 16 * (HH * CZ);
  const float* Wb = Wout + tn * 16;
  v8f acc = {};
  for (int k0 = 0; k0 < HH * CH; k0 += 32) {             // o part (192)
    v16bf af = make_frag([&](int m, int k) { return Ob[(long)m * (HH * CH) + k0 + k]; });
    v16bf bf = make_frag([&](int n, int k) { return Wb[(long)(k0 + k) * CS + n]; });
    acc = wmma_bf16(af, bf, acc);
  }
  for (int k0 = 0; k0 < HH * CZ; k0 += 32) {             // o_pair part (1536)
    v16bf af = make_frag([&](int m, int k) { return Pb[(long)m * (HH * CZ) + k0 + k]; });
    v16bf bf = make_frag([&](int n, int k) { return Wb[(long)(HH * CH + k0 + k) * CS + n]; });
    acc = wmma_bf16(af, bf, acc);
  }
  float* Cb = s_out + (long)tm * 16 * CS + tn * 16;
  store_acc(acc, [&](int m, int n, float v) { Cb[(long)m * CS + n] = v + bout[tn * 16 + n]; });
}

// ---------------------------------------------------------------------------
// Host launcher
// ---------------------------------------------------------------------------
extern "C" void kernel_launch(void* const* d_in, const int* in_sizes, int n_in,
                              void* d_out, int out_size, void* d_ws, size_t ws_size,
                              hipStream_t stream) {
  const float* s    = (const float*)d_in[0];
  const float* z    = (const float*)d_in[1];
  // d_in[2] = r : unused by reference
  const float* mask = (const float*)d_in[3];
  const float* Wq   = (const float*)d_in[4];
  const float* Wkv  = (const float*)d_in[5];
  const float* Wb   = (const float*)d_in[6];
  const float* Wout = (const float*)d_in[7];
  const float* bout = (const float*)d_in[8];

  float* out = (float*)d_out;
  float* out_s  = out;                                    // (B,N,C_S)
  float* out_a  = out + (long)BB * NN * CS;               // (B,H,N,N) softmax
  float* out_sd = out_a  + (long)BB * HH * NN * NN;       // a_sd
  float* out_ab = out_sd + (long)BB * HH * NN * NN;       // a_b

  float* ws   = (float*)d_ws;
  float* qws  = ws;                                       // 1024*192
  float* kvws = qws  + (long)BB * NN * HH * CH;           // 1024*384
  float* btws = kvws + (long)BB * NN * 2 * HH * CH;       // bt -> logits, B*H*N*N
  float* ows  = btws + (long)BB * HH * NN * NN;           // 1024*192
  float* opws = ows  + (long)BB * NN * HH * CH;           // 1024*1536

  const int M = BB * NN;                                  // 1024

  // K1: q and kv projections
  gemm_proj_kernel<<<(M / 16) * ((HH * CH) / 16), 32, 0, stream>>>(
      s, Wq, qws, HH * CH, CS, CS, HH * CH, HH * CH);
  gemm_proj_kernel<<<(M / 16) * ((2 * HH * CH) / 16), 32, 0, stream>>>(
      s, Wkv, kvws, 2 * HH * CH, CS, CS, 2 * HH * CH, 2 * HH * CH);

  // K2: pair bias bt (+ a_b output); 8 waves x 16 rows = 128 z rows per block
  bias_kernel<<<(int)(((long)BB * NN * NN) / 128), 256, 0, stream>>>(
      z, Wb, mask, btws, out_ab);

  // K3: attention scores + a_sd output + logits (in place over bt)
  logits_kernel<<<BB * HH * 32 * 32, 32, 0, stream>>>(qws, kvws, mask, btws, out_sd);

  // K4: softmax -> a output (also consumed by K5/K6)
  softmax_kernel<<<(BB * HH * NN) / 8, 256, 0, stream>>>(btws, out_a);

  // K5: o = a @ v
  av_kernel<<<BB * HH * 32, 32, 0, stream>>>(out_a, kvws, ows);

  // K6: o_pair = a @ z (TDM-staged z, 8 col tiles per block)
  az_kernel<<<BB * NN, 256, 0, stream>>>(out_a, z, opws);

  // K7: s_out = concat(o, o_pair) @ Wout + bout
  out_kernel<<<(M / 16) * (CS / 16), 32, 0, stream>>>(ows, opws, Wout, bout, out_s);
}
